// GraphConvolution_79121887527623
// MI455X (gfx1250) — matrix operand order repaired and checked by reading, not compiled, not run on hardware
//
#include <hip/hip_runtime.h>
#include <math.h>

// GCN layer for MI455X (gfx1250, wave32, WMMA).
//   N=8192, FIN=FOUT=256.
// Roofline: A@support streams 268MB of f32 adjacency (11.5us @ 23.3TB/s);
// bf16 WMMA (16x16x32, f32 accum) keeps the 34.4 GFLOP GEMM (~7us @ ~5 dense
// bf16 PFLOPS) under that bound. f32 WMMA (K=4) would be ~8x slower compute.

#define N_NODES 8192
#define FEAT    256
#define KCHUNK  32
#define NCHUNKS (N_NODES / KCHUNK)   // 256
#define ROWS_PER_WG 128
#define AGG_THREADS 256              // 8 waves x 32 lanes

typedef __attribute__((ext_vector_type(16))) __bf16 v16bf;
typedef __attribute__((ext_vector_type(8)))  float  v8f;
typedef __attribute__((ext_vector_type(4)))  float  f4;
typedef __attribute__((ext_vector_type(4)))  unsigned int u4;

// ---------------------------------------------------------------------------
// Kernel 1: D[i] = rsqrt(1 + sum_j adj[i][j])   (one row per block)
// ---------------------------------------------------------------------------
__global__ void gcn_degree(const float* __restrict__ adj, float* __restrict__ D) {
  __shared__ float red[256];
  const int row = blockIdx.x;
  const f4* rp = (const f4*)(adj + (size_t)row * N_NODES);
  float s = 0.f;
#pragma unroll
  for (int p = 0; p < 8; ++p) {
    f4 v = __builtin_nontemporal_load(&rp[threadIdx.x + p * 256]);
    s += v.x + v.y + v.z + v.w;
  }
  red[threadIdx.x] = s;
  __syncthreads();
  for (int off = 128; off > 0; off >>= 1) {
    if (threadIdx.x < off) red[threadIdx.x] += red[threadIdx.x + off];
    __syncthreads();
  }
  if (threadIdx.x == 0) D[row] = rsqrtf(1.0f + red[0]);
}

// ---------------------------------------------------------------------------
// Kernel 2: support = X @ W (f32, kept for the self-loop term), and
// bfrag = bf16(D_j * support) scattered into WMMA B-fragment order:
//   halfword index = ((kchunk*16 + n/16)*32 + (n%16) + 16*((k%16)/8))*16 + (k%8) + 8*((k%32)/16)
// One row j per block, one output feature f per thread.
// ---------------------------------------------------------------------------
__global__ void gcn_support(const float* __restrict__ x, const float* __restrict__ W,
                            const float* __restrict__ D,
                            float* __restrict__ support,
                            unsigned short* __restrict__ bfrag) {
  const int j = blockIdx.x;
  const int f = threadIdx.x;
  const float* xr = x + (size_t)j * FEAT;
  float s = 0.f;
#pragma unroll 8
  for (int k = 0; k < FEAT; ++k)
    s = fmaf(xr[k], W[(size_t)k * FEAT + f], s);
  support[(size_t)j * FEAT + f] = s;

  const float dj = D[j];
  const __bf16 bv = (__bf16)(dj * s);
  const int kk   = j & 31;
  const int c    = j >> 5;
  const int tile = f >> 4;
  const int lane = (f & 15) + (((kk >> 3) & 1) << 4);
  const int h    = (kk & 7) + ((kk >> 4) << 3);
  const size_t idx = (((size_t)c * 16 + tile) * 32 + lane) * (size_t)16 + h;
  bfrag[idx] = __builtin_bit_cast(unsigned short, bv);
}

// ---------------------------------------------------------------------------
// Kernel 3: out = relu(D_i * (adj @ bfrag) + D_i^2 * support[i] + b)
// 64 WGs x 8 waves; wave owns 16 rows x 256 cols (16 accum tiles).
// B chunks (32 x 256 bf16 = 16KB) double-buffered through LDS.
// adj read exactly once, converted f32->bf16 in-register into the A layout.
// ---------------------------------------------------------------------------
__global__ void __launch_bounds__(AGG_THREADS)
gcn_aggregate(const float* __restrict__ adj, const unsigned short* __restrict__ bfrag,
              const float* __restrict__ support, const float* __restrict__ D,
              const float* __restrict__ bias, float* __restrict__ out) {
  __shared__ unsigned int ldsB[2][4096];   // 2 x 16KB

  const int tid  = threadIdx.x;
  const int wave = tid >> 5;
  const int lane = tid & 31;
  const int r0   = blockIdx.x * ROWS_PER_WG + wave * 16;

  const v8f vzero = {0.f, 0.f, 0.f, 0.f, 0.f, 0.f, 0.f, 0.f};
  v8f acc[16];
#pragma unroll
  for (int t = 0; t < 16; ++t) acc[t] = vzero;

  const u4* bsrc = (const u4*)bfrag;   // one K-chunk = 1024 uint4 (16KB)

  // stage chunk 0
#pragma unroll
  for (int p = 0; p < 4; ++p)
    ((u4*)&ldsB[0][0])[p * AGG_THREADS + tid] = bsrc[p * AGG_THREADS + tid];
  __syncthreads();

  // A-fragment addressing (16x32 bf16 layout, ISA 7.12.2):
  //   lane<16 : row r0+lane,    K = kc+{0..7, 16..23}
  //   lane>=16: row r0+lane-16, K = kc+{8..15,24..31}
  const int   arow  = r0 + (lane & 15);
  const float* arowp = adj + (size_t)arow * N_NODES;
  const int   aoff  = (lane >> 4) * 8;

  for (int c = 0; c < NCHUNKS; ++c) {
    const int kc = c * KCHUNK;

    // prefetch next B chunk into registers (global -> reg)
    u4 pf[4];
    if (c + 1 < NCHUNKS) {
#pragma unroll
      for (int p = 0; p < 4; ++p)
        pf[p] = bsrc[(size_t)(c + 1) * 1024 + p * AGG_THREADS + tid];
    }

    // load 16 adjacency f32 per lane (whole 128B line consumed per row), cvt->bf16
    f4 a0 = __builtin_nontemporal_load((const f4*)(arowp + kc + aoff));
    f4 a1 = __builtin_nontemporal_load((const f4*)(arowp + kc + aoff + 4));
    f4 a2 = __builtin_nontemporal_load((const f4*)(arowp + kc + 16 + aoff));
    f4 a3 = __builtin_nontemporal_load((const f4*)(arowp + kc + 16 + aoff + 4));
    v16bf a;
    a[0]  = (__bf16)a0.x; a[1]  = (__bf16)a0.y; a[2]  = (__bf16)a0.z; a[3]  = (__bf16)a0.w;
    a[4]  = (__bf16)a1.x; a[5]  = (__bf16)a1.y; a[6]  = (__bf16)a1.z; a[7]  = (__bf16)a1.w;
    a[8]  = (__bf16)a2.x; a[9]  = (__bf16)a2.y; a[10] = (__bf16)a2.z; a[11] = (__bf16)a2.w;
    a[12] = (__bf16)a3.x; a[13] = (__bf16)a3.y; a[14] = (__bf16)a3.z; a[15] = (__bf16)a3.w;

    const unsigned int* bufp = &ldsB[c & 1][0];
#pragma unroll
    for (int t = 0; t < 16; ++t) {
      // B fragment: 32 contiguous bytes per lane (pre-swizzled), 2x ds_load_b128
      const u4* bp = (const u4*)(bufp + t * 256 + lane * 8);
      union { u4 u[2]; v16bf v; } cvt;
      cvt.u[0] = bp[0];
      cvt.u[1] = bp[1];
      acc[t] = __builtin_amdgcn_wmma_f32_16x16x32_bf16(
          /*neg_a=*/false, a, /*neg_b=*/false, cvt.v,
          /*c_mod=*/(short)0, acc[t], /*reuse_a=*/false, /*reuse_b=*/false);
    }

    // commit prefetched chunk to the other LDS buffer, then one barrier/iter
    if (c + 1 < NCHUNKS) {
#pragma unroll
      for (int p = 0; p < 4; ++p)
        ((u4*)&ldsB[(c + 1) & 1][0])[p * AGG_THREADS + tid] = pf[p];
    }
    __syncthreads();
  }

  // epilogue: out = relu(D_i*acc + D_i^2*support[i] + b)
  // C/D tile layout: vgpr v, lane<16 -> (M=v, N=lane); lane>=16 -> (M=v+8, N=lane-16)
  const int rbase = r0 + (lane >> 4) * 8;
  float dv[8];
#pragma unroll
  for (int v = 0; v < 8; ++v) dv[v] = D[rbase + v];
  const int colb = lane & 15;
#pragma unroll
  for (int t = 0; t < 16; ++t) {
    const int col = t * 16 + colb;
    const float bb = bias[col];
#pragma unroll
    for (int v = 0; v < 8; ++v) {
      const int row = rbase + v;
      float val = dv[v] * acc[t][v]
                + dv[v] * dv[v] * support[(size_t)row * FEAT + col]
                + bb;
      out[(size_t)row * FEAT + col] = val > 0.f ? val : 0.f;
    }
  }
}

// ---------------------------------------------------------------------------
extern "C" void kernel_launch(void* const* d_in, const int* in_sizes, int n_in,
                              void* d_out, int out_size, void* d_ws, size_t ws_size,
                              hipStream_t stream) {
  const float* x   = (const float*)d_in[0];   // [8192,256]
  const float* adj = (const float*)d_in[1];   // [8192,8192]
  const float* W   = (const float*)d_in[2];   // [256,256]
  const float* b   = (const float*)d_in[3];   // [256]
  float* out = (float*)d_out;                 // [8192,256]

  char* ws = (char*)d_ws;
  float* D                 = (float*)ws;                                   // 32 KB
  unsigned short* bfragbuf = (unsigned short*)(ws + 32 * 1024);            // 4 MB
  float* support           = (float*)(ws + 32 * 1024 + 4 * 1024 * 1024);   // 8 MB

  gcn_degree   <<<N_NODES, 256, 0, stream>>>(adj, D);
  gcn_support  <<<N_NODES, 256, 0, stream>>>(x, W, D, support, bfragbuf);
  gcn_aggregate<<<N_NODES / ROWS_PER_WG, AGG_THREADS, 0, stream>>>(
      adj, bfragbuf, support, D, b, out);
}
